// RPN_88819923681514
// MI455X (gfx1250) — matrix operand order repaired and checked by reading, not compiled, not run on hardware
//
#include <hip/hip_runtime.h>
#include <hip/hip_bf16.h>

// ---------------------------------------------------------------------------
// MI455X (gfx1250) RPN forward.
// Heavy compute: implicit-GEMM 3x3/1x1 conv via v_wmma_f32_16x16x32_bf16,
// f32 accumulate, bf16 conversion on the fly, LDS-staged 64x64 tiles,
// LDS-transposed coalesced epilogue, global_prefetch for next K-slab.
// Post-processing: softmax/anchor-decode, bitonic top-k, bitmask NMS.
// ---------------------------------------------------------------------------

typedef __attribute__((ext_vector_type(16))) __bf16 v16bf;
typedef __attribute__((ext_vector_type(8)))  __bf16 v8bf;
typedef __attribute__((ext_vector_type(8)))  float  v8f;

#define A_STRIDE 40          // 32 K-elems + 8 pad (bf16) -> conflict-free rows
#define C_STRIDE 65          // 64 + 1 pad (f32) -> conflict-free columns
#define NMS_N    6000
#define NMS_W    94          // ceil(6000/64)
#define POST_N   300
#define SORT_N   65536
#define NANCH    25

__device__ __forceinline__ unsigned short f32_to_bf16(float f) {
    unsigned int u = __float_as_uint(f);
    u += 0x7fffu + ((u >> 16) & 1u);       // round-to-nearest-even
    return (unsigned short)(u >> 16);
}

// ---------------- weight pack: f32 OIHW -> bf16 [Npad][Kpad], zero padded ----
__global__ void pack_w_kernel(const float* __restrict__ w, unsigned short* __restrict__ bp,
                              int Ktot, int Kpad, int Nact, int Npad) {
    int i = blockIdx.x * blockDim.x + threadIdx.x;
    int tot = Npad * Kpad;
    if (i >= tot) return;
    int k = i % Kpad, n = i / Kpad;
    float v = (k < Ktot && n < Nact) ? w[(size_t)n * Ktot + k] : 0.f;
    bp[i] = f32_to_bf16(v);
}

// ---------------- implicit-GEMM conv (pad=1 3x3 or 1x1), bias (+ReLU) -------
// GEMM: M = H*W spatial, N = Cout, K = Cin*ks*ks.  Block = 4 waves, 64x64 tile.
__global__ __launch_bounds__(128) void conv_wmma_kernel(
    const float* __restrict__ in, const unsigned short* __restrict__ Bp,
    const float* __restrict__ bias, float* __restrict__ out,
    int H, int W, int Ktot, int Kpad, int Nact, int k3, int relu)
{
    __shared__ unsigned short As[64 * A_STRIDE];  // [m][k]
    __shared__ unsigned short Bs[64 * A_STRIDE];  // [n][k]
    __shared__ float          Cs[64 * C_STRIDE];  // [m][n] f32 tile for epilogue

    const int M  = H * W;
    const int HW = M;
    const int m0 = blockIdx.x * 64;
    const int n0 = blockIdx.y * 64;
    const int t    = threadIdx.x;
    const int lane = t & 31;
    const int wid  = t >> 5;
    const int wrow = wid >> 1, wcol = wid & 1;
    const int lm    = lane & 15;
    const int khalf = lane >> 4;

    v8f acc00 = {}, acc01 = {}, acc10 = {}, acc11 = {};

    const int lrow = t >> 1;            // 0..63 tile row handled by this thread
    const int kb   = (t & 1) * 16;      // k sub-chunk

    // precompute spatial coords for A loading
    const int m_mine = m0 + lrow;
    const bool mok = (m_mine < M);
    int y0 = 0, x0 = 0;
    if (mok) { y0 = m_mine / W; x0 = m_mine - y0 * W; }

    const unsigned short* bsrc0 = Bp + (size_t)(n0 + lrow) * Kpad + kb;

    for (int k0 = 0; k0 < Kpad; k0 += 32) {
        // ---- stage A tile (im2col gather, f32 -> bf16), incremental indices
        if (k3) {
            int kg  = k0 + kb;
            int cin = kg / 9;
            int r   = kg - cin * 9;
            #pragma unroll
            for (int j = 0; j < 16; ++j) {
                float v = 0.f;
                if (mok && (kg + j) < Ktot) {
                    int ry = r / 3;
                    int y = y0 + ry - 1;
                    int x = x0 + (r - ry * 3) - 1;
                    if (y >= 0 && y < H && x >= 0 && x < W)
                        v = in[(size_t)cin * HW + y * W + x];
                }
                As[lrow * A_STRIDE + kb + j] = f32_to_bf16(v);
                if (++r == 9) { r = 0; ++cin; }
            }
        } else {
            #pragma unroll
            for (int j = 0; j < 16; ++j) {
                int kg = k0 + kb + j;
                float v = 0.f;
                if (mok && kg < Ktot)
                    v = in[(size_t)kg * HW + y0 * W + x0];
                As[lrow * A_STRIDE + kb + j] = f32_to_bf16(v);
            }
        }
        // ---- stage B tile: two b128 load/store pairs (16B-aligned both ends)
        {
            const unsigned short* src = bsrc0 + k0;
            *(uint4*)(void*)&Bs[lrow * A_STRIDE + kb]     = *(const uint4*)(const void*)(src);
            *(uint4*)(void*)&Bs[lrow * A_STRIDE + kb + 8] = *(const uint4*)(const void*)(src + 8);
            if (k0 + 32 < Kpad)
                __builtin_prefetch((const void*)(src + 32), 0, 1);   // global_prefetch_b8
        }
        __syncthreads();

        // ---- fragments per CDNA5 WMMA VGPR layouts ----
        const unsigned short* ar0 = &As[(wrow * 32 + lm)      * A_STRIDE];
        const unsigned short* ar1 = &As[(wrow * 32 + 16 + lm) * A_STRIDE];
        const unsigned short* br0 = &Bs[(wcol * 32 + lm)      * A_STRIDE];
        const unsigned short* br1 = &Bs[(wcol * 32 + 16 + lm) * A_STRIDE];

        v8bf a0lo = *(const v8bf*)(const void*)(ar0 + khalf * 8);
        v8bf a0hi = *(const v8bf*)(const void*)(ar0 + 16 + khalf * 8);
        v8bf a1lo = *(const v8bf*)(const void*)(ar1 + khalf * 8);
        v8bf a1hi = *(const v8bf*)(const void*)(ar1 + 16 + khalf * 8);
        v8bf b0lo = *(const v8bf*)(const void*)(br0 + khalf * 8);
        v8bf b0hi = *(const v8bf*)(const void*)(br0 + 16 + khalf * 8);
        v8bf b1lo = *(const v8bf*)(const void*)(br1 + khalf * 8);
        v8bf b1hi = *(const v8bf*)(const void*)(br1 + 16 + khalf * 8);

        v16bf a0 = __builtin_shufflevector(a0lo, a0hi, 0,1,2,3,4,5,6,7,8,9,10,11,12,13,14,15);
        v16bf a1 = __builtin_shufflevector(a1lo, a1hi, 0,1,2,3,4,5,6,7,8,9,10,11,12,13,14,15);
        v16bf b0 = __builtin_shufflevector(b0lo, b0hi, 0,1,2,3,4,5,6,7,8,9,10,11,12,13,14,15);
        v16bf b1 = __builtin_shufflevector(b1lo, b1hi, 0,1,2,3,4,5,6,7,8,9,10,11,12,13,14,15);

        acc00 = __builtin_amdgcn_wmma_f32_16x16x32_bf16(false, a0, false, b0, (short)0, acc00, false, false);
        acc01 = __builtin_amdgcn_wmma_f32_16x16x32_bf16(false, a0, false, b1, (short)0, acc01, false, false);
        acc10 = __builtin_amdgcn_wmma_f32_16x16x32_bf16(false, a1, false, b0, (short)0, acc10, false, false);
        acc11 = __builtin_amdgcn_wmma_f32_16x16x32_bf16(false, a1, false, b1, (short)0, acc11, false, false);

        __syncthreads();
    }

    // ---- epilogue: LDS transpose -> coalesced f32 NCHW stores ---------------
    {
        auto wr = [&](const v8f& a, int mt, int nt) {
            int nloc  = wcol * 32 + nt + lm;
            int mbase = wrow * 32 + mt + 8 * khalf;
            #pragma unroll
            for (int j = 0; j < 8; ++j)
                Cs[(mbase + j) * C_STRIDE + nloc] = a[j];
        };
        wr(acc00, 0,  0);
        wr(acc01, 0,  16);
        wr(acc10, 16, 0);
        wr(acc11, 16, 16);
    }
    __syncthreads();
    {
        int ml = t & 63;            // m within tile (contiguous across a wave)
        int nh = t >> 6;            // 0/1: two n-columns per iteration
        int mm = m0 + ml;
        #pragma unroll 4
        for (int nlo = 0; nlo < 64; nlo += 2) {
            int nloc = nlo + nh;
            int nn = n0 + nloc;
            if (nn < Nact && mm < M) {
                float v = Cs[ml * C_STRIDE + nloc] + bias[nn];
                if (relu) v = fmaxf(v, 0.f);
                out[(size_t)nn * M + mm] = v;
            }
        }
    }
}

// ---------------- 2x2 maxpool, NCHW ----------------------------------------
__global__ void maxpool_kernel(const float* __restrict__ in, float* __restrict__ out,
                               int C, int H, int W) {
    int Hn = H >> 1, Wn = W >> 1;
    int tot = C * Hn * Wn;
    int i = blockIdx.x * blockDim.x + threadIdx.x;
    if (i >= tot) return;
    int x = i % Wn; int y = (i / Wn) % Hn; int c = i / (Wn * Hn);
    const float* p = in + (size_t)c * H * W + (2 * y) * W + 2 * x;
    out[i] = fmaxf(fmaxf(p[0], p[1]), fmaxf(p[W], p[W + 1]));
}

// ---------------- base anchors ---------------------------------------------
__global__ void anchors_kernel(float* base) {
    int i = threadIdx.x;
    if (i >= NANCH) return;
    float tt = (float)i / 24.f;
    float s = expf(logf(2.f)    + tt * (logf(64.f) - logf(2.f)));
    float r = expf(logf(0.25f)  + tt * (logf(4.f)  - logf(0.25f)));
    float size = 16.f * s;
    float w = size / sqrtf(r);
    float h = w * r;
    const float c = 7.5f;
    base[i * 4 + 0] = c - (w - 1.f) * 0.5f;
    base[i * 4 + 1] = c - (h - 1.f) * 0.5f;
    base[i * 4 + 2] = c + (w - 1.f) * 0.5f;
    base[i * 4 + 3] = c + (h - 1.f) * 0.5f;
}

// ---------------- softmax + anchor decode + clip + min-size -----------------
__global__ void proposal_kernel(const float* __restrict__ score, const float* __restrict__ bbox,
                                const float* __restrict__ base, const float* __restrict__ im_info,
                                float* __restrict__ boxes, float* __restrict__ scores,
                                int H, int W) {
    int P = H * W * NANCH;
    int i = blockIdx.x * blockDim.x + threadIdx.x;
    if (i >= P) return;
    int a = i % NANCH;
    int m = i / NANCH;
    int hh = m / W, ww = m - hh * W;
    int HW = H * W;

    float s0 = score[a * HW + m];
    float s1 = score[(NANCH + a) * HW + m];
    float fg = 1.f / (1.f + expf(s0 - s1));

    float sx = ww * 16.f, sy = hh * 16.f;
    float ax1 = base[a * 4 + 0] + sx, ay1 = base[a * 4 + 1] + sy;
    float ax2 = base[a * 4 + 2] + sx, ay2 = base[a * 4 + 3] + sy;
    float wa = ax2 - ax1 + 1.f, ha = ay2 - ay1 + 1.f;
    float cxa = ax1 + 0.5f * wa, cya = ay1 + 0.5f * ha;

    float d0 = bbox[(4 * a + 0) * HW + m];
    float d1 = bbox[(4 * a + 1) * HW + m];
    float d2 = bbox[(4 * a + 2) * HW + m];
    float d3 = bbox[(4 * a + 3) * HW + m];

    float cx = d0 * wa + cxa, cy = d1 * ha + cya;
    float pw = expf(d2) * wa, ph = expf(d3) * ha;
    float Wm1 = im_info[1] - 1.f, Hm1 = im_info[0] - 1.f;
    float x1 = fminf(fmaxf(cx - 0.5f * pw, 0.f), Wm1);
    float y1 = fminf(fmaxf(cy - 0.5f * ph, 0.f), Hm1);
    float x2 = fminf(fmaxf(cx + 0.5f * pw, 0.f), Wm1);
    float y2 = fminf(fmaxf(cy + 0.5f * ph, 0.f), Hm1);

    float ms = 16.f * im_info[2];
    bool big = (x2 - x1 + 1.f >= ms) && (y2 - y1 + 1.f >= ms);
    boxes[i * 4 + 0] = x1; boxes[i * 4 + 1] = y1;
    boxes[i * 4 + 2] = x2; boxes[i * 4 + 3] = y2;
    scores[i] = big ? fg : -1.0e9f;
}

// ---------------- bitonic top-k (descending key, index payload) -------------
__global__ void sort_init_kernel(const float* __restrict__ scores, float* __restrict__ key,
                                 int* __restrict__ idx, int P) {
    int i = blockIdx.x * blockDim.x + threadIdx.x;
    if (i >= SORT_N) return;
    key[i] = (i < P) ? scores[i] : -3.4e38f;
    idx[i] = i;
}

__global__ void bitonic_kernel(float* __restrict__ key, int* __restrict__ idx, int j, int k) {
    int i = blockIdx.x * blockDim.x + threadIdx.x;
    int ixj = i ^ j;
    if (ixj <= i || i >= SORT_N) return;
    float ki = key[i], kj = key[ixj];
    bool desc = ((i & k) == 0);
    bool sw = desc ? (ki < kj) : (ki > kj);
    if (sw) {
        key[i] = kj; key[ixj] = ki;
        int ti = idx[i]; idx[i] = idx[ixj]; idx[ixj] = ti;
    }
}

__global__ void gather_kernel(const float* __restrict__ boxes, const float* __restrict__ key,
                              const int* __restrict__ idx,
                              float* __restrict__ bk, float* __restrict__ sk) {
    int i = blockIdx.x * blockDim.x + threadIdx.x;
    if (i >= NMS_N) return;
    int g = idx[i];
    bk[i * 4 + 0] = boxes[g * 4 + 0];
    bk[i * 4 + 1] = boxes[g * 4 + 1];
    bk[i * 4 + 2] = boxes[g * 4 + 2];
    bk[i * 4 + 3] = boxes[g * 4 + 3];
    sk[i] = key[i];
}

// ---------------- NMS: parallel IoU bitmask --------------------------------
__global__ void nms_mask_kernel(const float* __restrict__ bk, unsigned long long* __restrict__ mask) {
    int gid = blockIdx.x * blockDim.x + threadIdx.x;
    int tot = NMS_N * NMS_W;
    if (gid >= tot) return;
    int i = gid / NMS_W, w = gid - i * NMS_W;
    float ax1 = bk[i * 4], ay1 = bk[i * 4 + 1], ax2 = bk[i * 4 + 2], ay2 = bk[i * 4 + 3];
    float areaA = (ax2 - ax1 + 1.f) * (ay2 - ay1 + 1.f);
    unsigned long long bits = 0ull;
    int jb = w * 64;
    for (int b = 0; b < 64; ++b) {
        int j = jb + b;
        if (j >= NMS_N || j == i) continue;
        float bx1 = bk[j * 4], by1 = bk[j * 4 + 1], bx2 = bk[j * 4 + 2], by2 = bk[j * 4 + 3];
        float areaB = (bx2 - bx1 + 1.f) * (by2 - by1 + 1.f);
        float xx1 = fmaxf(ax1, bx1), yy1 = fmaxf(ay1, by1);
        float xx2 = fminf(ax2, bx2), yy2 = fminf(ay2, by2);
        float inter = fmaxf(xx2 - xx1 + 1.f, 0.f) * fmaxf(yy2 - yy1 + 1.f, 0.f);
        float iou = inter / (areaA + areaB - inter);
        if (iou > 0.7f) bits |= (1ull << b);
    }
    mask[(size_t)i * NMS_W + w] = bits;
}

// ---------------- NMS: sequential scan + roi emit ---------------------------
__global__ void nms_seq_kernel(const float* __restrict__ bk, const float* __restrict__ sk,
                               const unsigned long long* __restrict__ mask,
                               float* __restrict__ rois) {
    __shared__ unsigned long long valid[NMS_W];
    __shared__ int flag;
    __shared__ int cnt;
    int t = threadIdx.x;
    for (int w = t; w < NMS_W; w += blockDim.x) {
        int base = w * 64;
        int nb = NMS_N - base;
        unsigned long long v;
        if (nb >= 64) v = ~0ull;
        else if (nb <= 0) v = 0ull;
        else v = (1ull << nb) - 1ull;
        valid[w] = v;
    }
    for (int j = t; j < POST_N * 6; j += blockDim.x) rois[j] = 0.f;
    if (t == 0) cnt = 0;
    __syncthreads();

    for (int i = 0; i < NMS_N; ++i) {
        if (t == 0) flag = (int)((valid[i >> 6] >> (i & 63)) & 1ull);
        __syncthreads();
        if (flag) {
            const unsigned long long* row = mask + (size_t)i * NMS_W;
            for (int w = t; w < NMS_W; w += blockDim.x) valid[w] &= ~row[w];
            if (t == 0 && sk[i] > -1.0e8f && cnt < POST_N) {
                rois[cnt * 6 + 0] = 0.f;
                rois[cnt * 6 + 1] = bk[i * 4 + 0];
                rois[cnt * 6 + 2] = bk[i * 4 + 1];
                rois[cnt * 6 + 3] = bk[i * 4 + 2];
                rois[cnt * 6 + 4] = bk[i * 4 + 3];
                rois[cnt * 6 + 5] = sk[i];
                cnt++;
            }
        }
        __syncthreads();
    }
}

// ===========================================================================
extern "C" void kernel_launch(void* const* d_in, const int* in_sizes, int n_in,
                              void* d_out, int out_size, void* d_ws, size_t ws_size,
                              hipStream_t stream) {
    (void)in_sizes; (void)n_in; (void)out_size; (void)ws_size;

    static const int CIN[13]  = {3, 64, 64, 128, 128, 256, 256, 256, 512, 512, 512, 512, 512};
    static const int COUT[13] = {64, 64, 128, 128, 256, 256, 256, 512, 512, 512, 512, 512, 512};

    const float* im_data = (const float*)d_in[32];
    const float* im_info = (const float*)d_in[33];

    // ---- carve workspace ----
    char* ws = (char*)d_ws;
    size_t off = 0;
    auto carve = [&](size_t bytes) -> void* {
        void* p = ws + off;
        off += (bytes + 255) & ~(size_t)255;
        return p;
    };
    const size_t ACT_ELEMS = 64ull * 608 * 1024;               // biggest activation
    float* actA = (float*)carve(ACT_ELEMS * sizeof(float));
    float* actB = (float*)carve(ACT_ELEMS * sizeof(float));
    unsigned short* wpack = (unsigned short*)carve(512ull * 4608 * sizeof(unsigned short));
    float* scoreBuf = (float*)carve(64ull  * 2432 * sizeof(float));
    float* bboxBuf  = (float*)carve(128ull * 2432 * sizeof(float));
    float* baseAnch = (float*)carve(NANCH * 4 * sizeof(float));
    float* boxes    = (float*)carve(60800ull * 4 * sizeof(float));
    float* scores   = (float*)carve(60800ull * sizeof(float));
    float* keys     = (float*)carve(SORT_N * sizeof(float));
    int*   idxs     = (int*)carve(SORT_N * sizeof(int));
    float* boxes_k  = (float*)carve(NMS_N * 4 * sizeof(float));
    float* scores_k = (float*)carve(NMS_N * sizeof(float));
    unsigned long long* nmsmask =
        (unsigned long long*)carve((size_t)NMS_N * NMS_W * sizeof(unsigned long long));

    // ---- VGG backbone ----
    const float* cur = im_data;
    float* bufs[2] = {actA, actB};
    int pb = 0;
    int H = 608, W = 1024;

    for (int i = 0; i < 13; ++i) {
        const float* wgt = (const float*)d_in[2 * i];
        const float* bia = (const float*)d_in[2 * i + 1];
        int Cin = CIN[i], Cout = COUT[i];
        int Ktot = Cin * 9;
        int Kpad = (Ktot + 31) / 32 * 32;
        int Npad = (Cout + 63) / 64 * 64;

        int ptot = Npad * Kpad;
        pack_w_kernel<<<(ptot + 255) / 256, 256, 0, stream>>>(wgt, wpack, Ktot, Kpad, Cout, Npad);

        float* outp = (i == 12) ? (float*)d_out : bufs[pb];
        int M = H * W;
        dim3 grid((M + 63) / 64, Npad / 64);
        conv_wmma_kernel<<<grid, 128, 0, stream>>>(cur, wpack, bia, outp,
                                                   H, W, Ktot, Kpad, Cout, 1, 1);
        cur = outp;
        if (outp == bufs[pb]) pb ^= 1;

        if (i == 1 || i == 3 || i == 6 || i == 9) {
            float* pout = bufs[pb];
            int Hn = H / 2, Wn = W / 2;
            int tp = Cout * Hn * Wn;
            maxpool_kernel<<<(tp + 255) / 256, 256, 0, stream>>>(cur, pout, Cout, H, W);
            cur = pout; pb ^= 1;
            H = Hn; W = Wn;
        }
    }
    // cur == feat == d_out, H=38, W=64, C=512
    const float* feat = (const float*)d_out;
    const int M = H * W;                       // 2432

    // ---- RPN conv 3x3 512->512 + ReLU ----
    {
        int Ktot = 512 * 9, Kpad = Ktot, Npad = 512;
        int ptot = Npad * Kpad;
        pack_w_kernel<<<(ptot + 255) / 256, 256, 0, stream>>>((const float*)d_in[26], wpack,
                                                              Ktot, Kpad, 512, Npad);
        dim3 grid((M + 63) / 64, Npad / 64);
        conv_wmma_kernel<<<grid, 128, 0, stream>>>(feat, wpack, (const float*)d_in[27], actA,
                                                   H, W, Ktot, Kpad, 512, 1, 1);
    }
    // ---- score head 1x1 512->50 ----
    {
        int Ktot = 512, Kpad = 512, Npad = 64;
        int ptot = Npad * Kpad;
        pack_w_kernel<<<(ptot + 255) / 256, 256, 0, stream>>>((const float*)d_in[28], wpack,
                                                              Ktot, Kpad, 50, Npad);
        dim3 grid((M + 63) / 64, Npad / 64);
        conv_wmma_kernel<<<grid, 128, 0, stream>>>(actA, wpack, (const float*)d_in[29], scoreBuf,
                                                   H, W, Ktot, Kpad, 50, 0, 0);
    }
    // ---- bbox head 1x1 512->100 ----
    {
        int Ktot = 512, Kpad = 512, Npad = 128;
        int ptot = Npad * Kpad;
        pack_w_kernel<<<(ptot + 255) / 256, 256, 0, stream>>>((const float*)d_in[30], wpack,
                                                              Ktot, Kpad, 100, Npad);
        dim3 grid((M + 63) / 64, Npad / 64);
        conv_wmma_kernel<<<grid, 128, 0, stream>>>(actA, wpack, (const float*)d_in[31], bboxBuf,
                                                   H, W, Ktot, Kpad, 100, 0, 0);
    }

    // ---- proposals ----
    anchors_kernel<<<1, 32, 0, stream>>>(baseAnch);
    const int P = M * NANCH;                   // 60800
    proposal_kernel<<<(P + 255) / 256, 256, 0, stream>>>(scoreBuf, bboxBuf, baseAnch, im_info,
                                                         boxes, scores, H, W);

    // ---- top-6000 via global bitonic sort (descending) ----
    sort_init_kernel<<<SORT_N / 256, 256, 0, stream>>>(scores, keys, idxs, P);
    for (int k = 2; k <= SORT_N; k <<= 1)
        for (int j = k >> 1; j > 0; j >>= 1)
            bitonic_kernel<<<SORT_N / 256, 256, 0, stream>>>(keys, idxs, j, k);
    gather_kernel<<<(NMS_N + 255) / 256, 256, 0, stream>>>(boxes, keys, idxs, boxes_k, scores_k);

    // ---- NMS ----
    {
        int tot = NMS_N * NMS_W;
        nms_mask_kernel<<<(tot + 255) / 256, 256, 0, stream>>>(boxes_k, nmsmask);
    }
    float* rois = (float*)d_out + 512 * 38 * 64;
    nms_seq_kernel<<<1, 128, 0, stream>>>(boxes_k, scores_k, nmsmask, rois);
}
